// MultiScaleSetAbstraction_29953101922675
// MI455X (gfx1250) — matrix operand order repaired and checked by reading, not compiled, not run on hardware
//
#include <hip/hip_runtime.h>

typedef __attribute__((ext_vector_type(16))) _Float16 v16h;
typedef __attribute__((ext_vector_type(8)))  _Float16 v8h;
typedef __attribute__((ext_vector_type(8)))  float    v8f;

#define NB     8
#define NPTS   4096
#define NCENT  1024
#define NGROUP (NB * NCENT)   // 8192
#define IN_CH  64

// ---------------------------------------------------------------------------
// Farthest point sampling: one block per batch, 1024 threads, 4 pts/thread.
// Sequential over 1024 centroids; block-wide argmax reduction in LDS.
// ---------------------------------------------------------------------------
__global__ __launch_bounds__(1024) void fps_kernel(const float* __restrict__ xyz,
                                                   int* __restrict__ cent)
{
    __shared__ float sd[1024];
    __shared__ int   si[1024];
    const int b = blockIdx.x;
    const int t = threadIdx.x;
    const float* xb = xyz + (size_t)b * NPTS * 3;

    float dist[4];
#pragma unroll
    for (int i = 0; i < 4; i++) dist[i] = 1e10f;

    int cur = 0;
    for (int k = 0; k < NCENT; k++) {
        if (t == 0) cent[b * NCENT + k] = cur;
        const float cx = xb[cur * 3 + 0];
        const float cy = xb[cur * 3 + 1];
        const float cz = xb[cur * 3 + 2];
        float best = -1.0f; int bi = 0x7fffffff;
#pragma unroll
        for (int i = 0; i < 4; i++) {
            const int j = i * 1024 + t;
            const float dx = xb[j * 3 + 0] - cx;
            const float dy = xb[j * 3 + 1] - cy;
            const float dz = xb[j * 3 + 2] - cz;
            const float d = dx * dx + dy * dy + dz * dz;
            dist[i] = fminf(dist[i], d);
            if (dist[i] > best || (dist[i] == best && j < bi)) { best = dist[i]; bi = j; }
        }
        sd[t] = best; si[t] = bi;
        __syncthreads();
        for (int s = 512; s > 0; s >>= 1) {
            if (t < s) {
                const float od = sd[t + s]; const int oi = si[t + s];
                if (od > sd[t] || (od == sd[t] && oi < si[t])) { sd[t] = od; si[t] = oi; }
            }
            __syncthreads();
        }
        cur = si[0];
        __syncthreads();
    }
}

// ---------------------------------------------------------------------------
// Gather new_xyz = xyz[centroids]  -> d_out (first 8*1024*3 floats)
// ---------------------------------------------------------------------------
__global__ void newxyz_kernel(const float* __restrict__ xyz,
                              const int* __restrict__ cent,
                              float* __restrict__ out)
{
    const int b = blockIdx.x, s = threadIdx.x;
    const int idx = cent[b * NCENT + s];
    const float* p = xyz + ((size_t)b * NPTS + idx) * 3;
    float* o = out + ((size_t)b * NCENT + s) * 3;
    o[0] = p[0]; o[1] = p[1]; o[2] = p[2];
}

// ---------------------------------------------------------------------------
// Ball query + gather + pad to 96 ch (f16). One wave per group (4 waves/blk).
// Reference keeps the first `ns` in-radius indices in ascending order, filling
// with the first hit. Wave-cooperative scan with ballot prefix counts.
// ---------------------------------------------------------------------------
__global__ __launch_bounds__(128) void group_kernel(const float* __restrict__ xyz,
                                                    const float* __restrict__ points,
                                                    const float* __restrict__ newxyz,
                                                    _Float16* __restrict__ Xg,
                                                    int ns, float r2)
{
    __shared__ int gidx[4][64];
    const int lane = threadIdx.x & 31;
    const int w    = threadIdx.x >> 5;
    const int g    = blockIdx.x * 4 + w;
    const int b    = g >> 10;
    const float* xb = xyz + (size_t)b * NPTS * 3;

    const float px = newxyz[g * 3 + 0];
    const float py = newxyz[g * 3 + 1];
    const float pz = newxyz[g * 3 + 2];

    int cnt = 0;
    for (int base = 0; base < NPTS; base += 32) {
        const int j = base + lane;
        const float dx = xb[j * 3 + 0] - px;
        const float dy = xb[j * 3 + 1] - py;
        const float dz = xb[j * 3 + 2] - pz;
        const float d2 = dx * dx + dy * dy + dz * dz;
        const bool in = (d2 <= r2);
        const unsigned m = __builtin_amdgcn_ballot_w32(in);
        if (in) {
            const int pos = cnt + __popc(m & ((1u << lane) - 1u));
            if (pos < ns) gidx[w][pos] = j;
        }
        cnt += __popc(m);
        if (cnt >= ns) break;                 // wave-uniform
    }
    __syncthreads();
    if (cnt < ns) {                            // fill with first hit (centroid always hits)
        const int first = (cnt > 0) ? gidx[w][0] : 0;
        for (int i = cnt + lane; i < ns; i += 32) gidx[w][i] = first;
    }
    __syncthreads();

    const float* pb = points + (size_t)b * NPTS * IN_CH;
    for (int s2 = 0; s2 < ns; s2++) {
        const int idx = gidx[w][s2];
        const float* psrc = pb + (size_t)idx * IN_CH;
        const float* q    = xb + (size_t)idx * 3;
        const size_t ro = ((size_t)g * ns + s2) * 96;
#pragma unroll
        for (int cc = 0; cc < 3; cc++) {
            const int c = lane + cc * 32;
            float v;
            if (c < 3)        v = q[c] - (c == 0 ? px : (c == 1 ? py : pz));
            else if (c < 67)  v = psrc[c - 3];
            else              v = 0.0f;
            Xg[ro + c] = (_Float16)v;
        }
    }
}

// ---------------------------------------------------------------------------
// Convert fp32 weights [Cout, Kin] -> f16 [Cout, Kpad] (zero-padded K)
// ---------------------------------------------------------------------------
__global__ void wconv_kernel(const float* __restrict__ w, _Float16* __restrict__ wf,
                             int Cout, int Kin, int Kpad)
{
    const int i = blockIdx.x * 256 + threadIdx.x;
    if (i >= Cout * Kpad) return;
    const int o = i / Kpad, k = i % Kpad;
    wf[i] = (_Float16)((k < Kin) ? w[(size_t)o * Kin + k] : 0.0f);
}

// ---------------------------------------------------------------------------
// WMMA GEMM: Y[M,C] = A[M,Kpad] * W[C,Kpad]^T   (f16 in, f32 accum, f16 out)
// Wave computes a 64x16 slab: 4 accumulators, v_wmma_f32_16x16x32_f16.
// Fragment layout per CDNA5 ISA 7.12.2 (16-bit A 16x32; B 32x16; 32-bit C/D).
// ---------------------------------------------------------------------------
union AF { v16h v; v8h h[2]; };

__global__ __launch_bounds__(128) void gemm_kernel(const _Float16* __restrict__ A,
                                                   const _Float16* __restrict__ W,
                                                   _Float16* __restrict__ Y,
                                                   int M, int C, int Kpad)
{
    const int lane = threadIdx.x & 31;
    const int wv   = threadIdx.x >> 5;
    const int half = lane >> 4;        // 0: lanes 0-15, 1: lanes 16-31
    const int l16  = lane & 15;
    const int m0   = (blockIdx.x * 4 + wv) * 64;
    const int n0   = blockIdx.y * 16;

    v8f cacc[4] = {};
    const _Float16* wrow = W + (size_t)(n0 + l16) * Kpad + 8 * half;
    const _Float16* arow = A + (size_t)(m0 + l16) * Kpad + 8 * half;

    for (int kk = 0; kk < Kpad; kk += 32) {
        AF bf;
        bf.h[0] = *(const v8h*)(wrow + kk);
        bf.h[1] = *(const v8h*)(wrow + kk + 16);
#pragma unroll
        for (int t = 0; t < 4; t++) {
            AF af;
            const _Float16* ap = arow + (size_t)(t * 16) * Kpad + kk;
            af.h[0] = *(const v8h*)(ap);
            af.h[1] = *(const v8h*)(ap + 16);
            cacc[t] = __builtin_amdgcn_wmma_f32_16x16x32_f16(
                false, af.v, false, bf.v, (short)0, cacc[t], false, false);
        }
    }
#pragma unroll
    for (int t = 0; t < 4; t++) {
#pragma unroll
        for (int v = 0; v < 8; v++) {
            const int row = m0 + t * 16 + v + 8 * half;
            Y[(size_t)row * C + n0 + l16] = (_Float16)cacc[t][v];
        }
    }
}

// ---------------------------------------------------------------------------
// BatchNorm statistics: deterministic two-stage reduction (no float atomics).
// ---------------------------------------------------------------------------
__global__ void stats_partial_kernel(const _Float16* __restrict__ Y, int M, int C,
                                     int NSL, float* __restrict__ psum,
                                     float* __restrict__ psq)
{
    const int c = threadIdx.x, s = blockIdx.x;
    float sum = 0.0f, sq = 0.0f;
    for (int r = s; r < M; r += NSL) {
        const float v = (float)Y[(size_t)r * C + c];
        sum += v; sq += v * v;
    }
    psum[s * C + c] = sum;
    psq [s * C + c] = sq;
}

__global__ void stats_final_kernel(const float* __restrict__ psum,
                                   const float* __restrict__ psq,
                                   int NSL, int C, float invM,
                                   const float* __restrict__ gamma,
                                   const float* __restrict__ beta,
                                   float* __restrict__ scale,
                                   float* __restrict__ shift)
{
    const int c = threadIdx.x;
    float s = 0.0f, q = 0.0f;
    for (int i = 0; i < NSL; i++) { s += psum[i * C + c]; q += psq[i * C + c]; }
    const float mean = s * invM;
    const float var  = q * invM - mean * mean;
    const float istd = rsqrtf(var + 1e-5f);
    const float gsc  = gamma[c] * istd;
    scale[c] = gsc;
    shift[c] = beta[c] - mean * gsc;
}

__global__ void bn_apply_kernel(_Float16* __restrict__ Y, size_t total, int C,
                                const float* __restrict__ scale,
                                const float* __restrict__ shift)
{
    const size_t i = (size_t)blockIdx.x * 256 + threadIdx.x;
    if (i >= total) return;
    const int c = (int)(i % (size_t)C);
    const float v = (float)Y[i] * scale[c] + shift[c];
    Y[i] = (_Float16)fmaxf(v, 0.0f);
}

// ---------------------------------------------------------------------------
// Max over nsample -> combined f16 [8192, 320] at channel offset.
// ---------------------------------------------------------------------------
__global__ void maxpool_kernel(const _Float16* __restrict__ Z, int ns, int C,
                               _Float16* __restrict__ comb, int coff)
{
    const int m = blockIdx.x;
    const int c = threadIdx.x;
    const _Float16* z = Z + (size_t)m * ns * C + c;
    float v = -1e30f;
    for (int j = 0; j < ns; j++) v = fmaxf(v, (float)z[(size_t)j * C]);
    comb[(size_t)m * 320 + coff + c] = (_Float16)v;
}

// ---------------------------------------------------------------------------
// Transpose-store fused output: out[b, c, s] (float32)
// ---------------------------------------------------------------------------
__global__ void writeout_kernel(const _Float16* __restrict__ Yf, float* __restrict__ out)
{
    const int s = blockIdx.x * 256 + threadIdx.x;  // 0..1023
    const int c = blockIdx.y;                      // 0..319
    const int b = blockIdx.z;                      // 0..7
    const float v = (float)Yf[((size_t)b * NCENT + s) * 320 + c];
    out[((size_t)b * 320 + c) * NCENT + s] = v;
}

// ---------------------------------------------------------------------------
extern "C" void kernel_launch(void* const* d_in, const int* in_sizes, int n_in,
                              void* d_out, int out_size, void* d_ws, size_t ws_size,
                              hipStream_t stream)
{
    (void)in_sizes; (void)n_in; (void)out_size; (void)ws_size;

    // Input order (insertion order): xyz, points, 3 branches x 3 layers x
    // {w,b,gamma,beta}, fusion_w, fusion_b, fusion_gamma, fusion_beta.
    const float* xyz          = (const float*)d_in[0];
    const float* points       = (const float*)d_in[1];
    const float* fusion_w     = (const float*)d_in[38];
    const float* fusion_gamma = (const float*)d_in[40];
    const float* fusion_beta  = (const float*)d_in[41];

    float* out_newxyz = (float*)d_out;
    float* out_fused  = (float*)d_out + (size_t)NB * NCENT * 3;

    size_t off = 0;
    auto wsa = [&](size_t bytes) -> void* {
        void* p = (char*)d_ws + off;
        off += (bytes + 255) & ~(size_t)255;
        return p;
    };
    _Float16* Xg   = (_Float16*)wsa((size_t)524288 * 96 * 2);   // grouped f16 (max branch)
    _Float16* Ya   = (_Float16*)wsa((size_t)524288 * 128 * 2);  // activations ping
    _Float16* Yb   = (_Float16*)wsa((size_t)524288 * 128 * 2);  // activations pong
    _Float16* Wf   = (_Float16*)wsa((size_t)320 * 320 * 2);     // f16 weight staging
    float*    psum = (float*)wsa((size_t)256 * 320 * 4);
    float*    psq  = (float*)wsa((size_t)256 * 320 * 4);
    float*    scl  = (float*)wsa(320 * 4);
    float*    shf  = (float*)wsa(320 * 4);
    int*      cent = (int*)wsa((size_t)NB * NCENT * 4);
    _Float16* comb = (_Float16*)wsa((size_t)NGROUP * 320 * 2);
    _Float16* Yf   = (_Float16*)wsa((size_t)NGROUP * 320 * 2);

    fps_kernel<<<NB, 1024, 0, stream>>>(xyz, cent);
    newxyz_kernel<<<NB, NCENT, 0, stream>>>(xyz, cent, out_newxyz);

    const int   NSAMP[3]  = {16, 32, 64};
    const float RAD[3]    = {0.1f, 0.2f, 0.4f};
    const int   MLP[3][3] = {{32, 32, 64}, {64, 64, 128}, {64, 96, 128}};
    const int   COFF[3]   = {0, 64, 192};

    for (int br = 0; br < 3; br++) {
        const int ns = NSAMP[br];
        const int M  = NGROUP * ns;
        group_kernel<<<NGROUP / 4, 128, 0, stream>>>(xyz, points, out_newxyz, Xg,
                                                     ns, RAD[br] * RAD[br]);
        const _Float16* Xcur = Xg;
        int Kin = 67, Kpad = 96;
        for (int l = 0; l < 3; l++) {
            const int Cout = MLP[br][l];
            const int base = 2 + br * 12 + l * 4;
            const float* w     = (const float*)d_in[base];      // bias absorbed by BN
            const float* gamma = (const float*)d_in[base + 2];
            const float* beta  = (const float*)d_in[base + 3];
            const int nW = Cout * Kpad;
            wconv_kernel<<<(nW + 255) / 256, 256, 0, stream>>>(w, Wf, Cout, Kin, Kpad);
            _Float16* Yl = (l & 1) ? Yb : Ya;
            gemm_kernel<<<dim3(M / 256, Cout / 16), 128, 0, stream>>>(Xcur, Wf, Yl,
                                                                      M, Cout, Kpad);
            stats_partial_kernel<<<256, Cout, 0, stream>>>(Yl, M, Cout, 256, psum, psq);
            stats_final_kernel<<<1, Cout, 0, stream>>>(psum, psq, 256, Cout,
                                                       1.0f / (float)M, gamma, beta,
                                                       scl, shf);
            const size_t total = (size_t)M * Cout;
            bn_apply_kernel<<<(unsigned)((total + 255) / 256), 256, 0, stream>>>(
                Yl, total, Cout, scl, shf);
            Xcur = Yl; Kin = Cout; Kpad = Cout;
        }
        maxpool_kernel<<<NGROUP, MLP[br][2], 0, stream>>>(Xcur, ns, MLP[br][2],
                                                          comb, COFF[br]);
    }

    // Fusion: [8192,320] x [320,320]^T, BN over rows, ReLU, transpose-out.
    wconv_kernel<<<(320 * 320 + 255) / 256, 256, 0, stream>>>(fusion_w, Wf, 320, 320, 320);
    gemm_kernel<<<dim3(NGROUP / 256, 20), 128, 0, stream>>>(comb, Wf, Yf, NGROUP, 320, 320);
    stats_partial_kernel<<<64, 320, 0, stream>>>(Yf, NGROUP, 320, 64, psum, psq);
    stats_final_kernel<<<1, 320, 0, stream>>>(psum, psq, 64, 320, 1.0f / (float)NGROUP,
                                              fusion_gamma, fusion_beta, scl, shf);
    bn_apply_kernel<<<(NGROUP * 320 + 255) / 256, 256, 0, stream>>>(
        Yf, (size_t)NGROUP * 320, 320, scl, shf);
    writeout_kernel<<<dim3(4, 320, NB), 256, 0, stream>>>(Yf, out_fused);
}
